// PowerPIGNN_20401094656034
// MI455X (gfx1250) — compile-verified, matrix operand order checked by reading, not run on hardware
//
// PIGNN forward for MI455X (gfx1250, wave32, WMMA).
//
// Every MLP (din -> 256 -> 128 -> 32, ReLU, training-mode BatchNorm on input)
// is fused into one WMMA kernel over 64-row tiles; hidden activations live in
// 48KB of LDS, never HBM.  GEMMs use v_wmma_f32_16x16x32_f16 (f32 accum).
// Weights are pre-packed into WMMA-B fragment order so each B fragment is a
// contiguous 32-byte per-lane load (2x global_load_b128); A fragments are two
// contiguous 16-byte LDS runs per lane (2x ds_load_b128) hoisted into
// registers and reused across all column tiles.  The kernel is templated on
// padded input width so all fragment arrays unroll into registers.
//
// d_in flatten order (top-level insertion order, params in JAX pytree order =
// sorted dict keys, tuples as (W, b)):
//   0 nf[50000x5] 1 ef[800000x3] 2 u[8x2] 3 edge_index[2x800000](int)
//   4 node_graph_id[50000](int) 5 edge_graph_id[800000](int)
//   then for gn[0..3]: (gn[3] first has att: W[32x1], b[1])
//     edge:  beta,gamma,(W0,b0),(W1,b1),(W2,b2)
//     global:beta,gamma,(W0,b0),(W1,b1),(W2,b2)
//     node:  beta,gamma,(W0,b0),(W1,b1),(W2,b2)
//   then reg: (W0 32x64,b0),(W1 64x32,b1),(W2 32x16,b2),(W3 16x1,b3)

#include <hip/hip_runtime.h>
#include <hip/hip_bf16.h>
#include <cstdint>

typedef __attribute__((ext_vector_type(16))) _Float16 v16h;
typedef __attribute__((ext_vector_type(2)))  _Float16 v2h;
typedef __attribute__((ext_vector_type(8)))  float    v8f;

#define E_TOT 800000
#define N_TOT 50000
#define B_TOT 8
#define BN_EPS 1e-5f

// -------------------------------------------------------------------------
// Concatenated-input descriptor: x[row] = [s0[row] | s1[i1[row]] | s2[i2[row]]
// | s3[i3[row]]]   (null index pointer means identity row index)
struct Cat {
  const float* s0; int d0;
  const float* s1; int d1; const int* i1;
  const float* s2; int d2; const int* i2;
  const float* s3; int d3; const int* i3;
};

__device__ __forceinline__ float cat_load(const Cat& c, int row, int col) {
  if (col < c.d0) return c.s0[(size_t)row * c.d0 + col];
  col -= c.d0;
  if (col < c.d1) { int r = c.i1 ? c.i1[row] : row; return c.s1[(size_t)r * c.d1 + col]; }
  col -= c.d1;
  if (col < c.d2) { int r = c.i2 ? c.i2[row] : row; return c.s2[(size_t)r * c.d2 + col]; }
  col -= c.d2;
  int r = c.i3 ? c.i3[row] : row;
  return c.s3[(size_t)r * c.d3 + col];
}

// -------------------------------------------------------------------------
// Utility fills (graph-capture safe)
__global__ void fill_f32(float* p, float v, size_t n) {
  size_t i = (size_t)blockIdx.x * blockDim.x + threadIdx.x;
  if (i < n) p[i] = v;
}
__global__ void fill_u32(unsigned* p, unsigned v, size_t n) {
  size_t i = (size_t)blockIdx.x * blockDim.x + threadIdx.x;
  if (i < n) p[i] = v;
}

// -------------------------------------------------------------------------
// BatchNorm statistics: per-column sum / sum-of-squares over all rows.
#define BN_ROWS 1024
__global__ __launch_bounds__(256)
void bn_stats_kernel(Cat c, int R, int din, float* __restrict__ gsum,
                     float* __restrict__ gsq) {
  __shared__ float ls[128];
  __shared__ float lq[128];
  int tid = threadIdx.x;
  for (int i = tid; i < din; i += 256) { ls[i] = 0.f; lq[i] = 0.f; }
  __syncthreads();
  int r0 = blockIdx.x * BN_ROWS;
  int nr = min(BN_ROWS, R - r0);
  for (int idx = tid; idx < nr * din; idx += 256) {
    int r = r0 + idx / din;
    int col = idx % din;
    float v = cat_load(c, r, col);
    atomicAdd(&ls[col], v);
    atomicAdd(&lq[col], v * v);
  }
  __syncthreads();
  for (int i = tid; i < din; i += 256) {
    atomicAdd(&gsum[i], ls[i]);
    atomicAdd(&gsq[i], lq[i]);
  }
}

__global__ void bn_finalize_kernel(const float* gsum, const float* gsq,
                                   const float* gamma, const float* beta,
                                   int din, float R, float* scale, float* shift) {
  int i = blockIdx.x * blockDim.x + threadIdx.x;
  if (i >= din) return;
  float mu = gsum[i] / R;
  float var = gsq[i] / R - mu * mu;
  float is = rsqrtf(var + BN_EPS);
  float sc = gamma[i] * is;
  scale[i] = sc;
  shift[i] = beta[i] - mu * sc;
}

// -------------------------------------------------------------------------
// WMMA fragment K-permutation (CDNA5 16-bit 16x32 A-matrix layout, mirrored
// for B): for lane group g (lane>>4), half h maps to
//   k = h<8 ? g*8 + h : 16 + g*8 + (h-8)            (two contiguous runs)
__device__ __host__ __forceinline__ int frag_k(int grp, int h) {
  return (h < 8) ? (grp * 8 + h) : (16 + grp * 8 + (h - 8));
}

// f32 weights [din x dout] -> f16 fragment-packed:
//   o[(((kcB * dout/16) + ct) * 32 + lane) * 16 + h] = W[(kcB*32 + k(h)) * dout + ct*16 + n16]
__global__ void pack_w_kernel(const float* __restrict__ w, _Float16* __restrict__ o,
                              int din, int din_pad, int dout) {
  int idx = blockIdx.x * blockDim.x + threadIdx.x;
  int total = din_pad * dout;
  if (idx >= total) return;
  int h = idx & 15;
  int lane = (idx >> 4) & 31;
  int tile = idx >> 9;
  int nct = dout >> 4;
  int kcB = tile / nct;
  int ct = tile - kcB * nct;
  int grp = lane >> 4, n = lane & 15;
  int k = kcB * 32 + frag_k(grp, h);
  o[idx] = (k < din) ? (_Float16)w[(size_t)k * dout + ct * 16 + n] : (_Float16)0.f;
}

union AF { v16h v; _Float16 h[16]; };
union CF { v8f  v; float     f[8]; };

// A fragment: two contiguous 16-byte runs (p and p+16 halves)
__device__ __forceinline__ v16h load_afrag(const _Float16* p) {
  AF a;
#pragma unroll
  for (int h = 0; h < 8; ++h) a.h[h] = p[h];
#pragma unroll
  for (int h = 0; h < 8; ++h) a.h[8 + h] = p[16 + h];
  return a.v;
}
// B fragment: one contiguous 32-byte run per lane from the packed image
__device__ __forceinline__ v16h load_bfrag(const _Float16* __restrict__ Wf, int tile,
                                           int lane) {
  AF b;
  const _Float16* p = Wf + (((size_t)tile * 32 + lane) << 4);
#pragma unroll
  for (int h = 0; h < 16; ++h) b.h[h] = p[h];
  return b.v;
}

// -------------------------------------------------------------------------
// Fused MLP: out = relu(relu(relu(BN(x)@W0+b0)@W1+b1)@W2+b2),  dout = 32.
// 64 rows / block, 8 waves.  Wave w owns row-tile rt = w>>1 and half of the
// column tiles; all K-chunk A-fragments are hoisted into registers.
template <int DINP>
__global__ __launch_bounds__(256)
void mlp_wmma_kernel(Cat c, int R,
                     const float* __restrict__ scale, const float* __restrict__ shift,
                     int din,
                     const _Float16* __restrict__ Wf0, const float* __restrict__ b0,
                     const _Float16* __restrict__ Wf1, const float* __restrict__ b1,
                     const _Float16* __restrict__ Wf2, const float* __restrict__ b2,
                     float* __restrict__ out) {
  __shared__ _Float16 sA[64 * 128];   // stage0: X (64 x DINP); stage2 out: H1 (64 x 128)
  __shared__ _Float16 sH0[64 * 256];  // stage1 out: H0
  const int tid = threadIdx.x;
  const int lane = tid & 31;
  const int wv = tid >> 5;
  const int n16 = lane & 15;
  const int grp = lane >> 4;
  const int rowBase = blockIdx.x * 64;
  constexpr int NKC0 = DINP / 32;

  // ---- stage 0: gather + BatchNorm + f16 convert (paired stores)
  for (int idx = tid; idx < 64 * (DINP / 2); idx += 256) {
    int r = idx / (DINP / 2);
    int col = (idx - r * (DINP / 2)) * 2;
    int rg = rowBase + r;
    float v0 = 0.f, v1 = 0.f;
    if (rg < R) {
      if (col < din) v0 = cat_load(c, rg, col) * scale[col] + shift[col];
      if (col + 1 < din) v1 = cat_load(c, rg, col + 1) * scale[col + 1] + shift[col + 1];
    }
    v2h pv = { (_Float16)v0, (_Float16)v1 };
    *(v2h*)&sA[r * DINP + col] = pv;
  }
  __syncthreads();

  const int rt = wv >> 1;

  // ---- stage 1: H0[64x256] = relu(X @ W0 + b0); 8 column tiles per wave
  {
    v16h a[NKC0];
#pragma unroll
    for (int kcB = 0; kcB < NKC0; ++kcB)
      a[kcB] = load_afrag(&sA[(rt * 16 + n16) * DINP + kcB * 32 + grp * 8]);
    const int ctBase = (wv & 1) * 8;
#pragma unroll
    for (int j = 0; j < 8; ++j) {
      int ct = ctBase + j;
      CF acc;
#pragma unroll
      for (int r = 0; r < 8; ++r) acc.f[r] = 0.f;
#pragma unroll
      for (int kcB = 0; kcB < NKC0; ++kcB)
        acc.v = __builtin_amdgcn_wmma_f32_16x16x32_f16(
            false, a[kcB], false, load_bfrag(Wf0, kcB * 16 + ct, lane),
            (short)0, acc.v, false, false);
      float bias = b0[ct * 16 + n16];
#pragma unroll
      for (int r = 0; r < 8; ++r) {
        int mm = rt * 16 + r + grp * 8;
        float v = acc.f[r] + bias;
        sH0[mm * 256 + ct * 16 + n16] = (_Float16)(v > 0.f ? v : 0.f);
      }
    }
  }
  __syncthreads();

  // ---- stage 2: H1[64x128] = relu(H0 @ W1 + b1) -> sA; 4 column tiles/wave
  {
    v16h a[8];
#pragma unroll
    for (int kcB = 0; kcB < 8; ++kcB)
      a[kcB] = load_afrag(&sH0[(rt * 16 + n16) * 256 + kcB * 32 + grp * 8]);
    const int ctBase = (wv & 1) * 4;
#pragma unroll
    for (int j = 0; j < 4; ++j) {
      int ct = ctBase + j;
      CF acc;
#pragma unroll
      for (int r = 0; r < 8; ++r) acc.f[r] = 0.f;
#pragma unroll
      for (int kcB = 0; kcB < 8; ++kcB)
        acc.v = __builtin_amdgcn_wmma_f32_16x16x32_f16(
            false, a[kcB], false, load_bfrag(Wf1, kcB * 8 + ct, lane),
            (short)0, acc.v, false, false);
      float bias = b1[ct * 16 + n16];
#pragma unroll
      for (int r = 0; r < 8; ++r) {
        int mm = rt * 16 + r + grp * 8;
        float v = acc.f[r] + bias;
        sA[mm * 128 + ct * 16 + n16] = (_Float16)(v > 0.f ? v : 0.f);
      }
    }
  }
  __syncthreads();

  // ---- stage 3: out[64x32] = relu(H1 @ W2 + b2) -> global; 1 tile per wave
  {
    v16h a[4];
#pragma unroll
    for (int kcB = 0; kcB < 4; ++kcB)
      a[kcB] = load_afrag(&sA[(rt * 16 + n16) * 128 + kcB * 32 + grp * 8]);
    const int ct = wv & 1;
    CF acc;
#pragma unroll
    for (int r = 0; r < 8; ++r) acc.f[r] = 0.f;
#pragma unroll
    for (int kcB = 0; kcB < 4; ++kcB)
      acc.v = __builtin_amdgcn_wmma_f32_16x16x32_f16(
          false, a[kcB], false, load_bfrag(Wf2, kcB * 2 + ct, lane),
          (short)0, acc.v, false, false);
    float bias = b2[ct * 16 + n16];
#pragma unroll
    for (int r = 0; r < 8; ++r) {
      int mm = rt * 16 + r + grp * 8;
      int rg = rowBase + mm;
      if (rg < R) {
        float v = acc.f[r] + bias;
        out[(size_t)rg * 32 + ct * 16 + n16] = v > 0.f ? v : 0.f;
      }
    }
  }
}

// -------------------------------------------------------------------------
// Segment sum of 32-wide rows + row count (for segment mean)
__global__ void seg_sum32_kernel(const float* __restrict__ x, const int* __restrict__ ids,
                                 int rows, float* __restrict__ sum, float* __restrict__ cnt) {
  size_t idx = (size_t)blockIdx.x * blockDim.x + threadIdx.x;
  if (idx >= (size_t)rows * 32) return;
  int r = (int)(idx >> 5);
  int c = (int)(idx & 31);
  int g = ids[r];
  atomicAdd(&sum[(size_t)g * 32 + c], x[idx]);
  if (c == 0) atomicAdd(&cnt[g], 1.f);
}
__global__ void seg_div32_kernel(float* __restrict__ sum, const float* __restrict__ cnt,
                                 int n) {
  size_t idx = (size_t)blockIdx.x * blockDim.x + threadIdx.x;
  if (idx >= (size_t)n * 32) return;
  sum[idx] /= fmaxf(cnt[idx >> 5], 1.f);
}

// -------------------------------------------------------------------------
// Softmax attention aggregation (last layer)
__device__ __forceinline__ unsigned fenc(float f) {
  unsigned u = __float_as_uint(f);
  return (u & 0x80000000u) ? ~u : (u | 0x80000000u);
}
__device__ __forceinline__ float fdec(unsigned u) {
  u = (u & 0x80000000u) ? (u & 0x7fffffffu) : ~u;
  return __uint_as_float(u);
}
__global__ void att_score_kernel(const float* __restrict__ e, int E,
                                 const float* __restrict__ aW, const float* __restrict__ ab,
                                 float* __restrict__ sc) {
  int i = blockIdx.x * blockDim.x + threadIdx.x;
  if (i >= E) return;
  float s = ab[0];
#pragma unroll
  for (int k = 0; k < 32; ++k) s += e[(size_t)i * 32 + k] * aW[k];
  sc[i] = s;
}
__global__ void att_max_kernel(const float* __restrict__ sc, const int* __restrict__ dst,
                               int E, unsigned* __restrict__ nmax) {
  int i = blockIdx.x * blockDim.x + threadIdx.x;
  if (i >= E) return;
  atomicMax(&nmax[dst[i]], fenc(sc[i]));
}
__global__ void att_exp_kernel(const float* __restrict__ sc, const int* __restrict__ dst,
                               int E, const unsigned* __restrict__ nmax,
                               float* __restrict__ ex, float* __restrict__ den) {
  int i = blockIdx.x * blockDim.x + threadIdx.x;
  if (i >= E) return;
  float m = fdec(nmax[dst[i]]);
  float e = __expf(sc[i] - m);
  ex[i] = e;
  atomicAdd(&den[dst[i]], e);
}
__global__ void att_scatter_kernel(const float* __restrict__ e, const float* __restrict__ ex,
                                   const float* __restrict__ den, const int* __restrict__ dst,
                                   int E, float* __restrict__ agg) {
  size_t idx = (size_t)blockIdx.x * blockDim.x + threadIdx.x;
  if (idx >= (size_t)E * 32) return;
  int r = (int)(idx >> 5);
  int c = (int)(idx & 31);
  int d = dst[r];
  float w = ex[r] / fmaxf(den[d], 1e-16f);
  atomicAdd(&agg[(size_t)d * 32 + c], e[idx] * w);
}

// -------------------------------------------------------------------------
// Global-feature MLP: only 8 rows (M < 16), plain VALU, one block.
__global__ __launch_bounds__(256)
void global_mlp_kernel(const float* __restrict__ u, int gdim,
                       const float* __restrict__ ge, const float* __restrict__ gn,
                       const float* __restrict__ gamma, const float* __restrict__ beta,
                       const float* __restrict__ W0, const float* __restrict__ b0,
                       const float* __restrict__ W1, const float* __restrict__ b1,
                       const float* __restrict__ W2, const float* __restrict__ b2,
                       float* __restrict__ out) {
  const int din = gdim + 64;
  __shared__ float gx[8 * 96];
  __shared__ float h0[8 * 256];
  __shared__ float h1[8 * 128];
  int tid = threadIdx.x;
  for (int idx = tid; idx < 8 * din; idx += 256) {
    int r = idx / din, c = idx - r * din;
    float v;
    if (c < gdim) v = u[r * gdim + c];
    else if (c < gdim + 32) v = ge[r * 32 + (c - gdim)];
    else v = gn[r * 32 + (c - gdim - 32)];
    gx[r * din + c] = v;
  }
  __syncthreads();
  if (tid < din) {
    float mu = 0.f;
    for (int r = 0; r < 8; ++r) mu += gx[r * din + tid];
    mu *= 0.125f;
    float var = 0.f;
    for (int r = 0; r < 8; ++r) { float d = gx[r * din + tid] - mu; var += d * d; }
    var *= 0.125f;
    float sc = gamma[tid] * rsqrtf(var + BN_EPS);
    float sh = beta[tid] - mu * sc;
    for (int r = 0; r < 8; ++r) gx[r * din + tid] = gx[r * din + tid] * sc + sh;
  }
  __syncthreads();
  for (int idx = tid; idx < 8 * 256; idx += 256) {
    int r = idx >> 8, o = idx & 255;
    float s = b0[o];
    for (int k = 0; k < din; ++k) s += gx[r * din + k] * W0[k * 256 + o];
    h0[idx] = s > 0.f ? s : 0.f;
  }
  __syncthreads();
  for (int idx = tid; idx < 8 * 128; idx += 256) {
    int r = idx >> 7, o = idx & 127;
    float s = b1[o];
    for (int k = 0; k < 256; ++k) s += h0[r * 256 + k] * W1[k * 128 + o];
    h1[idx] = s > 0.f ? s : 0.f;
  }
  __syncthreads();
  for (int idx = tid; idx < 8 * 32; idx += 256) {
    int r = idx >> 5, o = idx & 31;
    float s = b2[o];
    for (int k = 0; k < 128; ++k) s += h1[r * 128 + k] * W2[k * 32 + o];
    out[idx] = s > 0.f ? s : 0.f;
  }
}

// -------------------------------------------------------------------------
// Final regression MLP per node: 32 -> 64 -> 32 -> 16 -> 1, ReLU, clip[0,1].
__global__ __launch_bounds__(256)
void reg_mlp_kernel(const float* __restrict__ nf, int N,
                    const float* W0, const float* b0, const float* W1, const float* b1,
                    const float* W2, const float* b2, const float* W3, const float* b3,
                    float* __restrict__ out) {
  __shared__ float s[4737];
  int tid = threadIdx.x;
  for (int i = tid; i < 2048; i += 256) s[i] = W0[i];            // W0 32x64
  for (int i = tid; i < 64; i += 256) s[2048 + i] = b0[i];
  for (int i = tid; i < 2048; i += 256) s[2112 + i] = W1[i];     // W1 64x32
  for (int i = tid; i < 32; i += 256) s[4160 + i] = b1[i];
  for (int i = tid; i < 512; i += 256) s[4192 + i] = W2[i];      // W2 32x16
  for (int i = tid; i < 16; i += 256) s[4704 + i] = b2[i];
  for (int i = tid; i < 16; i += 256) s[4720 + i] = W3[i];       // W3 16x1
  if (tid == 0) s[4736] = b3[0];
  __syncthreads();
  int n = blockIdx.x * 256 + tid;
  if (n >= N) return;
  float x[32];
#pragma unroll
  for (int k = 0; k < 32; ++k) x[k] = nf[(size_t)n * 32 + k];
  float h0[64];
  for (int o = 0; o < 64; ++o) {
    float v = s[2048 + o];
    for (int k = 0; k < 32; ++k) v += x[k] * s[k * 64 + o];
    h0[o] = v > 0.f ? v : 0.f;
  }
  float h1[32];
  for (int o = 0; o < 32; ++o) {
    float v = s[4160 + o];
    for (int k = 0; k < 64; ++k) v += h0[k] * s[2112 + k * 32 + o];
    h1[o] = v > 0.f ? v : 0.f;
  }
  float h2[16];
  for (int o = 0; o < 16; ++o) {
    float v = s[4704 + o];
    for (int k = 0; k < 32; ++k) v += h1[k] * s[4192 + k * 16 + o];
    h2[o] = v > 0.f ? v : 0.f;
  }
  float y = s[4736];
  for (int k = 0; k < 16; ++k) y += h2[k] * s[4720 + k];
  y = y > 0.f ? y : 0.f;
  y = y < 1.f ? y : 1.f;
  out[n] = y;
}

__global__ void add_inplace_kernel(float* __restrict__ a, const float* __restrict__ b,
                                   size_t n) {
  size_t i = (size_t)blockIdx.x * blockDim.x + threadIdx.x;
  if (i < n) a[i] += b[i];
}

// =========================================================================
extern "C" void kernel_launch(void* const* d_in, const int* in_sizes, int n_in,
                              void* d_out, int out_size, void* d_ws, size_t ws_size,
                              hipStream_t stream) {
  (void)in_sizes; (void)n_in; (void)out_size; (void)ws_size;

  const float* nf0 = (const float*)d_in[0];
  const float* ef0 = (const float*)d_in[1];
  const float* u0  = (const float*)d_in[2];
  const int* eidx  = (const int*)d_in[3];
  const int* srcI  = eidx;
  const int* dstI  = eidx + E_TOT;
  const int* ngid  = (const int*)d_in[4];
  const int* egid  = (const int*)d_in[5];

  // ---- parameter unpacking (pytree order; see header comment)
  int pi = 6;
  auto nxt = [&]() { return (const float*)d_in[pi++]; };
  struct MLPp { const float *beta, *gamma, *W0, *b0, *W1, *b1, *W2, *b2; };
  auto grab = [&]() {
    MLPp m;
    m.beta = nxt(); m.gamma = nxt();
    m.W0 = nxt(); m.b0 = nxt(); m.W1 = nxt(); m.b1 = nxt(); m.W2 = nxt(); m.b2 = nxt();
    return m;
  };
  MLPp edgeP[4], globP[4], nodeP[4];
  const float *attW = nullptr, *attB = nullptr;
  for (int i = 0; i < 4; ++i) {
    if (i == 3) { attW = nxt(); attB = nxt(); }
    edgeP[i] = grab();
    globP[i] = grab();
    nodeP[i] = grab();
  }
  const float* rW0 = nxt(); const float* rb0 = nxt();
  const float* rW1 = nxt(); const float* rb1 = nxt();
  const float* rW2 = nxt(); const float* rb2 = nxt();
  const float* rW3 = nxt(); const float* rb3 = nxt();

  // ---- workspace carving (~230 MB total; all chunks 32B aligned)
  float* wp = (float*)d_ws;
  auto carve = [&](size_t n) { float* r = wp; wp += n; return r; };
  float* eBuf[2] = { carve((size_t)E_TOT * 32), carve((size_t)E_TOT * 32) };
  float* nBuf[2] = { carve((size_t)N_TOT * 32), carve((size_t)N_TOT * 32) };
  float* uBuf[2] = { carve(B_TOT * 32), carve(B_TOT * 32) };
  float* agg   = carve((size_t)N_TOT * 32);
  float* cntN  = carve(N_TOT);
  float* geS   = carve(B_TOT * 32);
  float* geC   = carve(B_TOT);
  float* gnS   = carve(B_TOT * 32);
  float* gnC   = carve(B_TOT);
  float* score = carve(E_TOT);
  float* exb   = carve(E_TOT);
  float* den   = carve(N_TOT);
  unsigned* nmax = (unsigned*)carve(N_TOT);
  float* bnSum = carve(128);
  float* bnSq  = carve(128);   // contiguous with bnSum: zero both with one fill
  float* bnScale = carve(128);
  float* bnShift = carve(128);
  _Float16* hw0 = (_Float16*)carve(128 * 256 / 2);
  _Float16* hw1 = (_Float16*)carve(256 * 128 / 2);
  _Float16* hw2 = (_Float16*)carve(128 * 32 / 2);

  auto FILLF = [&](float* p, float v, size_t n) {
    fill_f32<<<(unsigned)((n + 255) / 256), 256, 0, stream>>>(p, v, n);
  };
  auto FILLU = [&](unsigned* p, unsigned v, size_t n) {
    fill_u32<<<(unsigned)((n + 255) / 256), 256, 0, stream>>>(p, v, n);
  };
  auto pad32 = [](int d) { return (d + 31) & ~31; };

  // Runs one fused BN+MLP over `rows` rows of concatenated input `c`.
  auto run_mlp = [&](const Cat& c, int rows, const MLPp& P, int din, float* outp) {
    int dinp = pad32(din);
    FILLF(bnSum, 0.f, 256);
    bn_stats_kernel<<<(rows + BN_ROWS - 1) / BN_ROWS, 256, 0, stream>>>(
        c, rows, din, bnSum, bnSq);
    bn_finalize_kernel<<<1, 128, 0, stream>>>(bnSum, bnSq, P.gamma, P.beta, din,
                                              (float)rows, bnScale, bnShift);
    pack_w_kernel<<<(dinp * 256 + 255) / 256, 256, 0, stream>>>(P.W0, hw0, din, dinp, 256);
    pack_w_kernel<<<(256 * 128 + 255) / 256, 256, 0, stream>>>(P.W1, hw1, 256, 256, 128);
    pack_w_kernel<<<(128 * 32 + 255) / 256, 256, 0, stream>>>(P.W2, hw2, 128, 128, 32);
    dim3 g((rows + 63) / 64);
    switch (dinp) {
      case 32:
        mlp_wmma_kernel<32><<<g, 256, 0, stream>>>(c, rows, bnScale, bnShift, din,
                                                   hw0, P.b0, hw1, P.b1, hw2, P.b2, outp);
        break;
      case 64:
        mlp_wmma_kernel<64><<<g, 256, 0, stream>>>(c, rows, bnScale, bnShift, din,
                                                   hw0, P.b0, hw1, P.b1, hw2, P.b2, outp);
        break;
      case 96:
        mlp_wmma_kernel<96><<<g, 256, 0, stream>>>(c, rows, bnScale, bnShift, din,
                                                   hw0, P.b0, hw1, P.b1, hw2, P.b2, outp);
        break;
      default:
        mlp_wmma_kernel<128><<<g, 256, 0, stream>>>(c, rows, bnScale, bnShift, din,
                                                    hw0, P.b0, hw1, P.b1, hw2, P.b2, outp);
        break;
    }
  };

  const float* eCur = ef0; int edim = 3;
  const float* nCur = nf0; int ndim = 5;
  const float* uCur = u0;  int gdim = 2;

  for (int i = 0; i < 4; ++i) {
    float* eOut = eBuf[i & 1];
    float* nOut = nBuf[i & 1];
    float* uOut = uBuf[i & 1];

    // ---- edge update: [ef | nf[src] | nf[dst] | u[egid]]
    Cat ce = { eCur, edim, nCur, ndim, srcI, nCur, ndim, dstI, uCur, gdim, egid };
    run_mlp(ce, E_TOT, edgeP[i], edim + 2 * ndim + gdim, eOut);

    // ---- edge aggregation per destination node
    FILLF(agg, 0.f, (size_t)N_TOT * 32);
    if (i < 3) {
      FILLF(cntN, 0.f, N_TOT);
      seg_sum32_kernel<<<(unsigned)(((size_t)E_TOT * 32 + 255) / 256), 256, 0, stream>>>(
          eOut, dstI, E_TOT, agg, cntN);
      seg_div32_kernel<<<(unsigned)(((size_t)N_TOT * 32 + 255) / 256), 256, 0, stream>>>(
          agg, cntN, N_TOT);
    } else {
      FILLU(nmax, 0u, N_TOT);  // 0 encodes the most-negative float
      FILLF(den, 0.f, N_TOT);
      att_score_kernel<<<(E_TOT + 255) / 256, 256, 0, stream>>>(eOut, E_TOT, attW, attB, score);
      att_max_kernel<<<(E_TOT + 255) / 256, 256, 0, stream>>>(score, dstI, E_TOT, nmax);
      att_exp_kernel<<<(E_TOT + 255) / 256, 256, 0, stream>>>(score, dstI, E_TOT, nmax, exb, den);
      att_scatter_kernel<<<(unsigned)(((size_t)E_TOT * 32 + 255) / 256), 256, 0, stream>>>(
          eOut, exb, den, dstI, E_TOT, agg);
    }

    // ---- per-graph mean of e_upd (pre-residual)
    FILLF(geS, 0.f, B_TOT * 32);
    FILLF(geC, 0.f, B_TOT);
    seg_sum32_kernel<<<(unsigned)(((size_t)E_TOT * 32 + 255) / 256), 256, 0, stream>>>(
        eOut, egid, E_TOT, geS, geC);
    seg_div32_kernel<<<1, 256, 0, stream>>>(geS, geC, B_TOT);

    // ---- node update: [nf | agg | u[ngid]]
    Cat cn = { nCur, ndim, agg, 32, nullptr, nullptr, 0, nullptr, uCur, gdim, ngid };
    run_mlp(cn, N_TOT, nodeP[i], ndim + 32 + gdim, nOut);

    // ---- per-graph mean of n_upd (pre-residual)
    FILLF(gnS, 0.f, B_TOT * 32);
    FILLF(gnC, 0.f, B_TOT);
    seg_sum32_kernel<<<(unsigned)(((size_t)N_TOT * 32 + 255) / 256), 256, 0, stream>>>(
        nOut, ngid, N_TOT, gnS, gnC);
    seg_div32_kernel<<<1, 256, 0, stream>>>(gnS, gnC, B_TOT);

    // ---- global update: [u | mean_edges | mean_nodes], 8 rows, plain VALU
    global_mlp_kernel<<<1, 256, 0, stream>>>(
        uCur, gdim, geS, gnS, globP[i].gamma, globP[i].beta,
        globP[i].W0, globP[i].b0, globP[i].W1, globP[i].b1, globP[i].W2, globP[i].b2, uOut);

    // ---- residual (layers >= 1)
    if (i >= 1) {
      add_inplace_kernel<<<(unsigned)(((size_t)E_TOT * 32 + 255) / 256), 256, 0, stream>>>(
          eOut, eCur, (size_t)E_TOT * 32);
      add_inplace_kernel<<<(unsigned)(((size_t)N_TOT * 32 + 255) / 256), 256, 0, stream>>>(
          nOut, nCur, (size_t)N_TOT * 32);
      add_inplace_kernel<<<1, 256, 0, stream>>>(uOut, uCur, (size_t)B_TOT * 32);
    }

    eCur = eOut; edim = 32;
    nCur = nOut; ndim = 32;
    uCur = uOut; gdim = 32;
  }

  // ---- final regression head + clip
  reg_mlp_kernel<<<(N_TOT + 255) / 256, 256, 0, stream>>>(
      nCur, N_TOT, rW0, rb0, rW1, rb1, rW2, rb2, rW3, rb3, (float*)d_out);
}
